// mvPAM_Module_mask_cascade_chl_13314398618058
// MI455X (gfx1250) — matrix operand order repaired and checked by reading, not compiled, not run on hardware
//
#include <hip/hip_runtime.h>

typedef __attribute__((ext_vector_type(16))) _Float16 v16h;
typedef __attribute__((ext_vector_type(8)))  float    v8f;
typedef __attribute__((ext_vector_type(4)))  int      v4i;

#define TB 256
#define BM 64
#define BN 128
#define BK 32
#define LDA_PAD 40
#define LDB_PAD 40

// ---------- CDNA5 async global->LDS path (guarded; falls back to reg copy) ----------
#if __has_builtin(__builtin_amdgcn_global_load_async_to_lds_b128)
#define HAVE_ASYNC_LDS 1
typedef __attribute__((address_space(1))) v4i gv4i;  // global int4
typedef __attribute__((address_space(3))) v4i lv4i;  // LDS int4
#else
#define HAVE_ASYNC_LDS 0
#endif

__device__ __forceinline__ void asyncCopy16(const _Float16* g, _Float16* l) {
#if HAVE_ASYNC_LDS
  // GLOBAL_LOAD_ASYNC_TO_LDS_B128: 16 bytes/lane, memory -> LDS, tracked by ASYNCcnt
  __builtin_amdgcn_global_load_async_to_lds_b128((gv4i*)g, (lv4i*)l, 0, 0);
#else
  *(uint4*)l = *(const uint4*)g;
#endif
}

__device__ __forceinline__ void waitAsync() {
#if HAVE_ASYNC_LDS
#if __has_builtin(__builtin_amdgcn_s_wait_asynccnt)
  __builtin_amdgcn_s_wait_asynccnt(0);
#else
  asm volatile("s_wait_asynccnt 0x0" ::: "memory");
#endif
#endif
}

// ---------------- fp32 -> fp16 convert ----------------
__global__ __launch_bounds__(TB) void k_f2h(const float* __restrict__ in,
                                            _Float16* __restrict__ out, long n) {
  long i = (long)blockIdx.x * TB + threadIdx.x;
  if (i < n) out[i] = (_Float16)in[i];
}

// ------------- transpose convert: (R,C) fp32 -> (C,R) fp16 -------------
__global__ __launch_bounds__(TB) void k_f2ht(const float* __restrict__ in,
                                             _Float16* __restrict__ out, int R, int C) {
  long i = (long)blockIdx.x * TB + threadIdx.x;
  long n = (long)R * C;
  if (i >= n) return;
  int r = (int)(i / C), c = (int)(i % C);
  out[(long)c * R + r] = (_Float16)in[i];
}

// ------- im2col 3x3 SAME: fp32 (C,H,W) -> fp16 col (H*W, C*9) -------
__global__ __launch_bounds__(TB) void k_im2col(const float* __restrict__ in,
                                               _Float16* __restrict__ col,
                                               int Cch, int H, int W) {
  long K9 = (long)Cch * 9;
  long total = (long)H * W * K9;
  long i = (long)blockIdx.x * TB + threadIdx.x;
  if (i >= total) return;
  int k = (int)(i % K9);
  int p = (int)(i / K9);
  int ch = k / 9, rr = k % 9;
  int y = p / W + rr / 3 - 1;
  int x = p % W + rr % 3 - 1;
  float v = 0.f;
  if (y >= 0 && y < H && x >= 0 && x < W) v = in[((long)ch * H + y) * W + x];
  col[i] = (_Float16)v;
}

// ---------------- block reductions ----------------
__device__ __forceinline__ float blockReduceMax(float v) {
  __shared__ float s[TB];
  int t = threadIdx.x;
  s[t] = v; __syncthreads();
  for (int k = TB / 2; k > 0; k >>= 1) { if (t < k) s[t] = fmaxf(s[t], s[t + k]); __syncthreads(); }
  float r = s[0]; __syncthreads(); return r;
}
__device__ __forceinline__ float blockReduceMin(float v) {
  __shared__ float s[TB];
  int t = threadIdx.x;
  s[t] = v; __syncthreads();
  for (int k = TB / 2; k > 0; k >>= 1) { if (t < k) s[t] = fminf(s[t], s[t + k]); __syncthreads(); }
  float r = s[0]; __syncthreads(); return r;
}
__device__ __forceinline__ float blockReduceSum(float v) {
  __shared__ float s[TB];
  int t = threadIdx.x;
  s[t] = v; __syncthreads();
  for (int k = TB / 2; k > 0; k >>= 1) { if (t < k) s[t] = s[t] + s[t + k]; __syncthreads(); }
  float r = s[0]; __syncthreads(); return r;
}

// ------- fused 3-way (plain + 2 masked) row softmax, fp32 in -> fp16 out -------
__global__ __launch_bounds__(TB) void k_softmax3(const float* __restrict__ E,
                                                 const float* __restrict__ M1,
                                                 const float* __restrict__ M2,
                                                 _Float16* __restrict__ a0,
                                                 _Float16* __restrict__ a1,
                                                 _Float16* __restrict__ a2, int N) {
  int row = blockIdx.x;
  const float* e  = E  + (long)row * N;
  const float* m1 = M1 + (long)row * N;
  const float* m2 = M2 + (long)row * N;
  float x0 = -3.4e38f, x1 = -3.4e38f, x2 = -3.4e38f;
  for (int j = threadIdx.x; j < N; j += TB) {
    float v = e[j];
    x0 = fmaxf(x0, v);
    if (m1[j] > 0.f) x1 = fmaxf(x1, v);
    if (m2[j] > 0.f) x2 = fmaxf(x2, v);
  }
  x0 = blockReduceMax(x0); x1 = blockReduceMax(x1); x2 = blockReduceMax(x2);
  float s0 = 0.f, s1 = 0.f, s2 = 0.f;
  for (int j = threadIdx.x; j < N; j += TB) {
    float v = e[j];
    s0 += __expf(v - x0);
    if (m1[j] > 0.f) s1 += __expf(v - x1);
    if (m2[j] > 0.f) s2 += __expf(v - x2);
  }
  s0 = blockReduceSum(s0); s1 = blockReduceSum(s1); s2 = blockReduceSum(s2);
  float r0 = 1.f / s0, r1 = 1.f / s1, r2 = 1.f / s2;
  for (int j = threadIdx.x; j < N; j += TB) {
    float v = e[j];
    long o = (long)row * N + j;
    a0[o] = (_Float16)(__expf(v - x0) * r0);
    a1[o] = (_Float16)((m1[j] > 0.f) ? __expf(v - x1) * r1 : 0.f);
    a2[o] = (_Float16)((m2[j] > 0.f) ? __expf(v - x2) * r2 : 0.f);
  }
}

// ------- chl softmax: attn = softmax(rowmax - e) == exp(rowmin - e)/sum -------
__global__ __launch_bounds__(TB) void k_chlsm(const float* __restrict__ E,
                                              _Float16* __restrict__ A, int L) {
  int row = blockIdx.x;
  const float* e = E + (long)row * L;
  float mn = 3.4e38f;
  for (int j = threadIdx.x; j < L; j += TB) mn = fminf(mn, e[j]);
  mn = blockReduceMin(mn);
  float s = 0.f;
  for (int j = threadIdx.x; j < L; j += TB) s += __expf(mn - e[j]);
  s = blockReduceSum(s);
  float r = 1.f / s;
  for (int j = threadIdx.x; j < L; j += TB)
    A[(long)row * L + j] = (_Float16)(__expf(mn - e[j]) * r);
}

// ---------------- WMMA NT GEMM (async double-buffered LDS) ----------------
// C[M,N] = sum_k A[m,k] * Bt[n,k]   (A: MxK f16 row-major, Bt: NxK f16 row-major)
// mode 0: C = acc
// mode 1: C = acc + bRow[m]
// mode 2: C = relu(acc * sRow[m] + bRow[m])
// mode 3: C = gamma[0]*acc + resid[m*N+n]
// Requires: M % 64 == 0, N % 128 == 0, K % 32 == 0.
__global__ __launch_bounds__(TB) void k_gemm_nt(
    const _Float16* __restrict__ A, const _Float16* __restrict__ Bt,
    float* __restrict__ C, int M, int N, int K,
    const float* __restrict__ sRow, const float* __restrict__ bRow,
    const float* __restrict__ resid, const float* __restrict__ gammaP, int mode) {
  __shared__ _Float16 lA[2][BM][LDA_PAD];   // 2 x 5120 B
  __shared__ _Float16 lB[2][BN][LDB_PAD];   // 2 x 10240 B
  const int tid  = threadIdx.x;
  const int lane = tid & 31;
  const int wave = tid >> 5;       // 8 waves
  const int wm   = wave & 1;       // 0..1 -> M tile
  const int wn   = wave >> 1;      // 0..3 -> N tile
  const int half = lane >> 4;      // 0/1
  const int l16  = lane & 15;
  const int bm0  = blockIdx.y * BM;
  const int bn0  = blockIdx.x * BN;

  const int ar = tid >> 2, akc = (tid & 3) * 8;    // A tile: 64 rows x 32, 8 halves/thread
  const int br = tid >> 1, bkc = (tid & 1) * 16;   // B tile: 128 rows x 32, 16 halves/thread
  const _Float16* aptr = A  + (long)(bm0 + ar) * K + akc;
  const _Float16* bptr = Bt + (long)(bn0 + br) * K + bkc;

  // preload first tile into buffer 0
  asyncCopy16(aptr,     &lA[0][ar][akc]);
  asyncCopy16(bptr,     &lB[0][br][bkc]);
  asyncCopy16(bptr + 8, &lB[0][br][bkc + 8]);

  v8f acc[2][2] = {};
  int buf = 0;

  for (int k0 = 0; k0 < K; k0 += BK, buf ^= 1) {
    waitAsync();       // my async writes for buffer `buf` have landed in LDS
    __syncthreads();   // everyone's have; also: nobody still reads buf^1

    if (k0 + BK < K) { // prefetch next tile into the other buffer (overlaps compute)
      asyncCopy16(aptr + k0 + BK,     &lA[buf ^ 1][ar][akc]);
      asyncCopy16(bptr + k0 + BK,     &lB[buf ^ 1][br][bkc]);
      asyncCopy16(bptr + k0 + BK + 8, &lB[buf ^ 1][br][bkc + 8]);
    }

    v16h af[2], bfv[2];
#pragma unroll
    for (int i = 0; i < 2; ++i) {
      // A frag (16x32 f16): lanes 0-15 rows M=0..15 K={0..7,16..23}; lanes 16-31 K={8..15,24..31}
      const _Float16* s = &lA[buf][wm * 32 + i * 16 + l16][half * 8];
      ((uint4*)&af[i])[0] = *(const uint4*)s;
      ((uint4*)&af[i])[1] = *(const uint4*)(s + 16);
    }
#pragma unroll
    for (int j = 0; j < 2; ++j) {
      // B frag (32x16 f16): lane holds column n=l16, lanes 0-15 K={0..7,16..23}; lanes 16-31 K={8..15,24..31}
      const _Float16* s = &lB[buf][wn * 32 + j * 16 + l16][half * 8];
      ((uint4*)&bfv[j])[0] = *(const uint4*)s;
      ((uint4*)&bfv[j])[1] = *(const uint4*)(s + 16);
    }
#pragma unroll
    for (int i = 0; i < 2; ++i)
#pragma unroll
      for (int j = 0; j < 2; ++j)
        acc[i][j] = __builtin_amdgcn_wmma_f32_16x16x32_f16(
            false, af[i], false, bfv[j], (short)0, acc[i][j], false, false);
  }

  float gamma = (mode == 3) ? gammaP[0] : 0.f;
#pragma unroll
  for (int i = 0; i < 2; ++i)
#pragma unroll
    for (int j = 0; j < 2; ++j) {
      int nIdx = bn0 + wn * 32 + j * 16 + l16;
      int mB   = bm0 + wm * 32 + i * 16 + 8 * half;  // C/D layout: M = r + 8*(lane/16)
#pragma unroll
      for (int r = 0; r < 8; ++r) {
        int m = mB + r;
        float v = acc[i][j][r];
        if (mode == 1)      v = v + bRow[m];
        else if (mode == 2) v = fmaxf(v * sRow[m] + bRow[m], 0.f);
        else if (mode == 3) v = gamma * v + resid[(long)m * N + nIdx];
        C[(long)m * N + nIdx] = v;
      }
    }
}

// ---------------- host orchestration ----------------
extern "C" void kernel_launch(void* const* d_in, const int* in_sizes, int n_in,
                              void* d_out, int out_size, void* d_ws, size_t ws_size,
                              hipStream_t stream) {
  (void)in_sizes; (void)n_in; (void)out_size; (void)ws_size;
  const int Bb = 4, Cc = 512, Hh = 48, Ww = 48, Nn = 2304, C4 = 128, K9 = 4608;

  const float* x      = (const float*)d_in[0];
  const float* wq     = (const float*)d_in[1];
  const float* bq     = (const float*)d_in[2];
  const float* wk     = (const float*)d_in[3];
  const float* bk     = (const float*)d_in[4];
  const float* wv0    = (const float*)d_in[5];
  const float* bv0    = (const float*)d_in[6];
  const float* c1w    = (const float*)d_in[7];
  const float* c1s    = (const float*)d_in[8];
  const float* c1b    = (const float*)d_in[9];
  const float* c2w    = (const float*)d_in[10];
  const float* c2s    = (const float*)d_in[11];
  const float* c2b    = (const float*)d_in[12];
  const float* gamma  = (const float*)d_in[13];
  const float* gamma1 = (const float*)d_in[14];
  const float* gamma2 = (const float*)d_in[15];
  const float* q1w    = (const float*)d_in[16];
  const float* q1s    = (const float*)d_in[17];
  const float* q1b    = (const float*)d_in[18];
  const float* e1w    = (const float*)d_in[19];
  const float* e1b    = (const float*)d_in[20];
  const float* q2w    = (const float*)d_in[21];
  const float* q2s    = (const float*)d_in[22];
  const float* q2b    = (const float*)d_in[23];
  const float* e2w    = (const float*)d_in[24];
  const float* e2b    = (const float*)d_in[25];
  const float* mask1  = (const float*)d_in[26];
  const float* mask2  = (const float*)d_in[27];
  float* out = (float*)d_out;

  char* wsb = (char*)d_ws;
  size_t off = 0;
  auto alloc = [&](size_t bytes) -> void* {
    void* p = wsb + off;
    off = (off + bytes + 255) & ~(size_t)255;
    return p;
  };

  // f16 weight copies (batch independent)
  _Float16* hwq  = (_Float16*)alloc((size_t)64  * Cc * 2);
  _Float16* hwk  = (_Float16*)alloc((size_t)64  * Cc * 2);
  _Float16* hwv  = (_Float16*)alloc((size_t)Cc  * Cc * 2);
  _Float16* hw1  = (_Float16*)alloc((size_t)Cc  * K9 * 2);
  _Float16* hw2  = (_Float16*)alloc((size_t)Cc  * K9 * 2);
  _Float16* hq1w = (_Float16*)alloc((size_t)C4  * Cc * 2);
  _Float16* hq2w = (_Float16*)alloc((size_t)C4  * Cc * 2);
  _Float16* he1w = (_Float16*)alloc((size_t)Cc  * C4 * 2);
  _Float16* he2w = (_Float16*)alloc((size_t)Cc  * C4 * 2);

  // per-batch scratch (reused across batches; stream is in-order)
  _Float16* xT   = (_Float16*)alloc((size_t)Nn * Cc * 2);
  float*    qf   = (float*)   alloc((size_t)64 * Nn * 4);
  float*    kf   = (float*)   alloc((size_t)64 * Nn * 4);
  _Float16* qT   = (_Float16*)alloc((size_t)Nn * 64 * 2);
  _Float16* kT   = (_Float16*)alloc((size_t)Nn * 64 * 2);
  float*    vf   = (float*)   alloc((size_t)Cc * Nn * 4);
  _Float16* vh   = (_Float16*)alloc((size_t)Cc * Nn * 2);
  // energy (NxN f32) and im2col (N x 4608 f16) are never live together: alias
  char*     eAlias = (char*)  alloc((size_t)Nn * Nn * 4);
  float*    energy = (float*)eAlias;
  _Float16* col    = (_Float16*)eAlias;
  _Float16* a0   = (_Float16*)alloc((size_t)Nn * Nn * 2);
  _Float16* a1   = (_Float16*)alloc((size_t)Nn * Nn * 2);
  _Float16* a2   = (_Float16*)alloc((size_t)Nn * Nn * 2);
  float*    resb = (float*)   alloc((size_t)Cc * Nn * 4);
  float*    of0  = (float*)   alloc((size_t)Cc * Nn * 4);
  _Float16* oh   = (_Float16*)alloc((size_t)Cc * Nn * 2);
  _Float16* ohT  = (_Float16*)alloc((size_t)Nn * Cc * 2);
  float*    qcf  = (float*)   alloc((size_t)C4 * Nn * 4);
  _Float16* qch  = (_Float16*)alloc((size_t)C4 * Nn * 2);
  float*    ecf  = (float*)   alloc((size_t)C4 * Cc * 4);
  _Float16* ecT  = (_Float16*)alloc((size_t)Cc * C4 * 2);
  float*    exf  = (float*)   alloc((size_t)Cc * Cc * 4);
  _Float16* attc = (_Float16*)alloc((size_t)Cc * Cc * 2);
  float*    v1f  = (float*)   alloc((size_t)Cc * Nn * 4);
  _Float16* v1h  = (_Float16*)alloc((size_t)Cc * Nn * 2);

  auto F2H = [&](const float* in, _Float16* o, long n) {
    k_f2h<<<dim3((unsigned)((n + TB - 1) / TB)), dim3(TB), 0, stream>>>(in, o, n);
  };
  auto F2HT = [&](const float* in, _Float16* o, int R, int Ccol) {
    long n = (long)R * Ccol;
    k_f2ht<<<dim3((unsigned)((n + TB - 1) / TB)), dim3(TB), 0, stream>>>(in, o, R, Ccol);
  };
  auto GEMM = [&](const _Float16* A, const _Float16* Bt, float* Cm, int M, int N, int K,
                  const float* s, const float* b, const float* res, const float* g, int mode) {
    k_gemm_nt<<<dim3(N / BN, M / BM), dim3(TB), 0, stream>>>(A, Bt, Cm, M, N, K, s, b, res, g, mode);
  };
  auto IM2COL = [&](const float* in) {
    long tot = (long)Nn * K9;
    k_im2col<<<dim3((unsigned)((tot + TB - 1) / TB)), dim3(TB), 0, stream>>>(in, col, Cc, Hh, Ww);
  };
  auto CHL = [&](const _Float16* qwH, const float* qs, const float* qb,
                 const _Float16* ewH, const float* eb) {
    GEMM(qwH, ohT, qcf, C4, Nn, Cc, qs, qb, nullptr, nullptr, 2);          // q = relu(bn(conv1x1))
    F2H(qcf, qch, (long)C4 * Nn);
    GEMM(qch, oh, ecf, C4, Cc, Nn, nullptr, nullptr, nullptr, nullptr, 0); // energy (C4 x C)
    F2HT(ecf, ecT, C4, Cc);
    GEMM(ewH, ecT, exf, Cc, Cc, C4, nullptr, eb, nullptr, nullptr, 1);     // expand + eb
    k_chlsm<<<dim3(Cc), dim3(TB), 0, stream>>>(exf, attc, Cc);             // softmax(max - e)
    GEMM(attc, ohT, v1f, Cc, Nn, Cc, nullptr, nullptr, nullptr, nullptr, 0); // attn @ k
    F2H(v1f, v1h, (long)Cc * Nn);
  };

  // weights -> f16
  F2H(wq,  hwq,  (long)64 * Cc);
  F2H(wk,  hwk,  (long)64 * Cc);
  F2H(wv0, hwv,  (long)Cc * Cc);
  F2H(c1w, hw1,  (long)Cc * K9);
  F2H(c2w, hw2,  (long)Cc * K9);
  F2H(q1w, hq1w, (long)C4 * Cc);
  F2H(q2w, hq2w, (long)C4 * Cc);
  F2H(e1w, he1w, (long)Cc * C4);
  F2H(e2w, he2w, (long)Cc * C4);

  for (int b = 0; b < Bb; ++b) {
    const float* xb = x + (size_t)b * Cc * Nn;
    float* outb = out + (size_t)b * Cc * Nn;

    F2HT(xb, xT, Cc, Nn);                                                  // x^T f16 (N x C)
    GEMM(hwq, xT, qf, 64, Nn, Cc, nullptr, bq, nullptr, nullptr, 1);       // q
    GEMM(hwk, xT, kf, 64, Nn, Cc, nullptr, bk, nullptr, nullptr, 1);       // k
    GEMM(hwv, xT, vf, Cc, Nn, Cc, nullptr, bv0, nullptr, nullptr, 1);      // v
    F2HT(qf, qT, 64, Nn);
    F2HT(kf, kT, 64, Nn);
    GEMM(qT, kT, energy, Nn, Nn, 64, nullptr, nullptr, nullptr, nullptr, 0); // energy NxN
    k_softmax3<<<dim3(Nn), dim3(TB), 0, stream>>>(energy, mask1, mask2, a0, a1, a2, Nn);
    F2H(vf, vh, (long)Cc * Nn);
    GEMM(vh, a0, resb, Cc, Nn, Nn, nullptr, nullptr, xb, gamma, 3);        // out0 = g*v@a0^T + x
    IM2COL(resb);                                                          // (overwrites energy)
    GEMM(hw1, col, of0, Cc, Nn, K9, c1s, c1b, nullptr, nullptr, 2);        // conv1 + bn + relu
    F2H(of0, oh, (long)Cc * Nn);
    F2HT(of0, ohT, Cc, Nn);

    CHL(hq1w, q1s, q1b, he1w, e1b);                                        // v1
    GEMM(v1h, a1, resb, Cc, Nn, Nn, nullptr, nullptr, of0, gamma1, 3);     // out1 pre-conv
    IM2COL(resb);
    GEMM(hw2, col, of0, Cc, Nn, K9, c2s, c2b, nullptr, nullptr, 2);        // conv2 + bn + relu
    F2H(of0, oh, (long)Cc * Nn);
    F2HT(of0, ohT, Cc, Nn);

    CHL(hq2w, q2s, q2b, he2w, e2b);                                        // v2
    GEMM(v1h, a2, outb, Cc, Nn, Nn, nullptr, nullptr, of0, gamma2, 3);     // final out
  }
}
